// SpatialAttentionModule_58978490908972
// MI455X (gfx1250) — matrix operand order repaired and checked by reading, not compiled, not run on hardware
//
#include <hip/hip_runtime.h>
#include <hip/hip_bf16.h>

typedef _Float16 v16h __attribute__((ext_vector_type(16)));
typedef _Float16 h8   __attribute__((ext_vector_type(8)));
typedef float    v8f  __attribute__((ext_vector_type(8)));

#define B_  4
#define C_  256
#define CQ  64
#define H_  64
#define W_  64
#define N_  4096   // H_*W_

// ---------------------------------------------------------------------------
// Kernel 1: q = conv1x3(x) + b, k = conv3x1(x) + b   ->  qT/kT [B][N][CQ] f16
// Block = one (b, h) row pair of outputs. 256 threads = 64 d  x 4 w-groups.
// ---------------------------------------------------------------------------
__global__ __launch_bounds__(256) void proj_qk_kernel(
    const float* __restrict__ x,
    const float* __restrict__ qw, const float* __restrict__ qb,
    const float* __restrict__ kw, const float* __restrict__ kb,
    _Float16* __restrict__ qT, _Float16* __restrict__ kT)
{
    __shared__ __align__(16) float xs[8 * 3 * 64]; // [cc][row(h-1,h,h+1)][w]
    const int bx  = blockIdx.x;
    const int b   = bx >> 6;
    const int h   = bx & 63;
    const int tid = threadIdx.x;
    const int d   = tid & 63;
    const int wg  = tid >> 6;
    const int wbase = wg * 16;

    float aq[16], ak[16];
    const float qbias = qb[d], kbias = kb[d];
#pragma unroll
    for (int t = 0; t < 16; ++t) { aq[t] = qbias; ak[t] = kbias; }

    for (int c0 = 0; c0 < C_; c0 += 8) {
        __syncthreads();
        for (int idx = tid; idx < 8 * 3 * 64; idx += 256) {
            const int cc  = idx / 192;
            const int rem = idx % 192;
            const int rr  = rem >> 6;
            const int w   = rem & 63;
            const int hh  = h - 1 + rr;
            float val = 0.0f;
            if (hh >= 0 && hh < H_)
                val = x[(((size_t)(b * C_ + c0 + cc) * H_ + hh) * W_) + w];
            xs[idx] = val;
        }
        __syncthreads();
#pragma unroll 2
        for (int cc = 0; cc < 8; ++cc) {
            const int c = c0 + cc;
            const float* wqp = qw + (size_t)(d * C_ + c) * 3;
            const float* wkp = kw + (size_t)(d * C_ + c) * 3;
            const float wq0 = wqp[0], wq1 = wqp[1], wq2 = wqp[2];
            const float wk0 = wkp[0], wk1 = wkp[1], wk2 = wkp[2];
            const float* xu = xs + cc * 192;      // row h-1
            const float* xm = xu + 64;            // row h
            const float* xd = xu + 128;           // row h+1
#pragma unroll
            for (int t = 0; t < 16; ++t) {
                const int w  = wbase + t;
                const float x0  = xm[w];
                const float xl  = (w > 0)  ? xm[w - 1] : 0.0f;
                const float xr  = (w < 63) ? xm[w + 1] : 0.0f;
                aq[t] += wq0 * xl + wq1 * x0 + wq2 * xr;
                ak[t] += wk0 * xu[w] + wk1 * x0 + wk2 * xd[w];
            }
        }
    }
#pragma unroll
    for (int t = 0; t < 16; ++t) {
        const int w = wbase + t;
        const int n = h * W_ + w;
        qT[((size_t)(b * N_ + n)) * CQ + d] = (_Float16)aq[t];
        kT[((size_t)(b * N_ + n)) * CQ + d] = (_Float16)ak[t];
    }
}

// ---------------------------------------------------------------------------
// Kernel 2: v = conv1x1(x) + b  ->  v [B][C][N] f16 (j-contiguous rows)
// Block = one (b, h). 256 threads, thread owns output channel c, 64 w accums.
// ---------------------------------------------------------------------------
__global__ __launch_bounds__(256) void proj_v_kernel(
    const float* __restrict__ x,
    const float* __restrict__ vw, const float* __restrict__ vb,
    _Float16* __restrict__ vh)
{
    __shared__ __align__(16) float xv[128 * 64]; // [cc][w]
    const int bx  = blockIdx.x;
    const int b   = bx >> 6;
    const int h   = bx & 63;
    const int tid = threadIdx.x;   // output channel c

    float acc[64];
    const float bias = vb[tid];
#pragma unroll
    for (int w = 0; w < 64; ++w) acc[w] = bias;

    for (int c0 = 0; c0 < C_; c0 += 128) {
        __syncthreads();
        for (int idx = tid; idx < 128 * 64; idx += 256) {
            const int cc = idx >> 6;
            const int w  = idx & 63;
            xv[idx] = x[(((size_t)(b * C_ + c0 + cc) * H_ + h) * W_) + w];
        }
        __syncthreads();
        for (int cc = 0; cc < 128; ++cc) {
            const float wv = vw[(size_t)tid * C_ + c0 + cc];
            const float* xr = xv + cc * 64;
#pragma unroll
            for (int w = 0; w < 64; ++w) acc[w] += wv * xr[w];
        }
    }
    _Float16* dst = vh + ((size_t)(b * C_ + tid)) * N_ + h * W_;
#pragma unroll
    for (int w = 0; w < 64; ++w) dst[w] = (_Float16)acc[w];
}

// ---------------------------------------------------------------------------
// Kernel 3: flash attention.  out = gamma * softmax(qT k) v + x
// Grid: B * (N/64) blocks of 128 threads (4 waves). Wave owns 16 query rows,
// accumulates O[16 x 256] in f32 via v_wmma_f32_16x16x32_f16, streaming
// 64-wide key/value tiles through LDS with online softmax.
// ---------------------------------------------------------------------------
__global__ __launch_bounds__(128) void attn_kernel(
    const _Float16* __restrict__ qT, const _Float16* __restrict__ kT,
    const _Float16* __restrict__ vh, const float* __restrict__ x,
    const float* __restrict__ gamma, float* __restrict__ out)
{
    __shared__ __align__(32) _Float16 Ksh[64 * 64];    //  8 KB  [jj][d]
    __shared__ __align__(32) _Float16 Vsh[256 * 64];   // 32 KB  [c][jj]
    __shared__ __align__(32) _Float16 Psh[4 * 16 * 64];//  8 KB  per-wave P

    const int b    = blockIdx.x >> 6;
    const int i0   = (blockIdx.x & 63) * 64;
    const int tid  = threadIdx.x;
    const int wv   = tid >> 5;
    const int lane = tid & 31;
    const int half = lane >> 4;   // 0: lanes 0-15 (rows r / K-lo), 1: lanes 16-31
    const int ln   = lane & 15;
    const int i0w  = i0 + wv * 16;

    // Q A-fragments (16x32 f16 each), d = 0..31 and 32..63
    v16h aQ[2];
#pragma unroll
    for (int s = 0; s < 2; ++s) {
        const _Float16* qrow = qT + ((size_t)(b * N_ + i0w + ln)) * CQ + 32 * s + (half ? 16 : 0);
        h8 c0 = *(const h8*)qrow;
        h8 c1 = *(const h8*)(qrow + 8);
        aQ[s] = __builtin_shufflevector(c0, c1, 0,1,2,3,4,5,6,7,8,9,10,11,12,13,14,15);
    }

    v8f o_[16];
#pragma unroll
    for (int ct = 0; ct < 16; ++ct) o_[ct] = (v8f){};
    float mrow[8], lrow[8];
#pragma unroll
    for (int r = 0; r < 8; ++r) { mrow[r] = -1e30f; lrow[r] = 0.0f; }

    const _Float16* kbase = kT + ((size_t)b * N_) * CQ;
    const _Float16* vbase = vh + ((size_t)b * C_) * N_;

    for (int jt = 0; jt < 64; ++jt) {
        const int j0 = jt * 64;
        __syncthreads();
        // stage K tile (contiguous 8KB run of kT) and V tile into LDS
        {
            const h8* src = (const h8*)(kbase + (size_t)j0 * CQ);
            h8* dst = (h8*)Ksh;
#pragma unroll
            for (int it = 0; it < 4; ++it) dst[it * 128 + tid] = src[it * 128 + tid];
        }
#pragma unroll
        for (int it = 0; it < 16; ++it) {
            const int q = it * 128 + tid;       // 2048 chunks of 8 f16
            const int c = q >> 3, part = q & 7;
            *(h8*)(Vsh + c * 64 + part * 8) =
                *(const h8*)(vbase + (size_t)c * N_ + j0 + part * 8);
        }
        if (jt + 1 < 64) {  // gfx1250 global_prefetch of next tiles
            __builtin_prefetch(kbase + (size_t)(j0 + 64) * CQ, 0, 0);
            __builtin_prefetch(vbase + (size_t)(lane) * N_ + j0 + 64, 0, 0);
        }
        __syncthreads();

        // S (16x64) = Q^T K : 4 column tiles x 2 K-steps
        v8f s_[4];
#pragma unroll
        for (int nt = 0; nt < 4; ++nt) {
            v8f s = (v8f){};
#pragma unroll
            for (int ss = 0; ss < 2; ++ss) {
                v16h bk = *(const v16h*)(Ksh + (nt * 16 + ln) * 64 + 32 * ss + (half ? 16 : 0));
                s = __builtin_amdgcn_wmma_f32_16x16x32_f16(
                        false, aQ[ss], false, bk, (short)0, s, false, false);
            }
            s_[nt] = s;
        }

        // online softmax over this j-tile; P -> per-wave LDS scratch (f16)
        float scale[8];
#pragma unroll
        for (int r = 0; r < 8; ++r) {
            float mx = s_[0][r];
            mx = fmaxf(mx, s_[1][r]); mx = fmaxf(mx, s_[2][r]); mx = fmaxf(mx, s_[3][r]);
#pragma unroll
            for (int msk = 1; msk < 16; msk <<= 1)
                mx = fmaxf(mx, __shfl_xor(mx, msk, 32));
            const float mn = fmaxf(mrow[r], mx);
            scale[r] = __expf(mrow[r] - mn);
            mrow[r]  = mn;
            float rs = 0.0f;
#pragma unroll
            for (int nt = 0; nt < 4; ++nt) {
                const float p = __expf(s_[nt][r] - mn);
                rs += p;
                Psh[(wv * 16 + r + (half ? 8 : 0)) * 64 + nt * 16 + ln] = (_Float16)p;
            }
#pragma unroll
            for (int msk = 1; msk < 16; msk <<= 1)
                rs += __shfl_xor(rs, msk, 32);
            lrow[r] = lrow[r] * scale[r] + rs;
        }
#pragma unroll
        for (int ct = 0; ct < 16; ++ct)
#pragma unroll
            for (int r = 0; r < 8; ++r) o_[ct][r] *= scale[r];

        // re-fragment P as A (same-wave LDS store->load is in-order)
        v16h aP[2];
#pragma unroll
        for (int ss = 0; ss < 2; ++ss) {
            const _Float16* prow = Psh + (wv * 16 + ln) * 64 + 32 * ss + (half ? 16 : 0);
            h8 c0 = *(const h8*)prow;
            h8 c1 = *(const h8*)(prow + 8);
            aP[ss] = __builtin_shufflevector(c0, c1, 0,1,2,3,4,5,6,7,8,9,10,11,12,13,14,15);
        }
        // O += P * V^T : 16 channel tiles x 2 K-steps
#pragma unroll
        for (int ct = 0; ct < 16; ++ct) {
#pragma unroll
            for (int ss = 0; ss < 2; ++ss) {
                v16h bv = *(const v16h*)(Vsh + (ct * 16 + ln) * 64 + 32 * ss + (half ? 16 : 0));
                o_[ct] = __builtin_amdgcn_wmma_f32_16x16x32_f16(
                            false, aP[ss], false, bv, (short)0, o_[ct], false, false);
            }
        }
    }

    // epilogue: out = gamma * O / l + x
    const float g = gamma[0];
    float invl[8];
#pragma unroll
    for (int r = 0; r < 8; ++r) invl[r] = 1.0f / lrow[r];
#pragma unroll
    for (int ct = 0; ct < 16; ++ct) {
        const int c = ct * 16 + ln;
#pragma unroll
        for (int r = 0; r < 8; ++r) {
            const int i = i0w + r + (half ? 8 : 0);
            const size_t idx = ((size_t)(b * C_ + c)) * N_ + i;
            out[idx] = g * (o_[ct][r] * invl[r]) + x[idx];
        }
    }
}

// ---------------------------------------------------------------------------
extern "C" void kernel_launch(void* const* d_in, const int* in_sizes, int n_in,
                              void* d_out, int out_size, void* d_ws, size_t ws_size,
                              hipStream_t stream) {
    (void)in_sizes; (void)n_in; (void)out_size; (void)ws_size;
    const float* x     = (const float*)d_in[0];
    const float* qw    = (const float*)d_in[1];
    const float* qb    = (const float*)d_in[2];
    const float* kw    = (const float*)d_in[3];
    const float* kb    = (const float*)d_in[4];
    const float* vw    = (const float*)d_in[5];
    const float* vb    = (const float*)d_in[6];
    const float* gamma = (const float*)d_in[7];
    float* out = (float*)d_out;

    _Float16* qT = (_Float16*)d_ws;                    // 2 MB
    _Float16* kT = qT + (size_t)B_ * N_ * CQ;          // 2 MB
    _Float16* vh = kT + (size_t)B_ * N_ * CQ;          // 8 MB

    proj_qk_kernel<<<B_ * H_, 256, 0, stream>>>(x, qw, qb, kw, kb, qT, kT);
    proj_v_kernel <<<B_ * H_, 256, 0, stream>>>(x, vw, vb, vh);
    attn_kernel   <<<B_ * (N_ / 64), 128, 0, stream>>>(qT, kT, vh, x, gamma, out);
}
